// DefaultHumanoidGRUCritic_51840255262918
// MI455X (gfx1250) — compile-verified, matrix operand order checked by reading, not run on hardware
//
#include <hip/hip_runtime.h>

// ---------------------------------------------------------------------------
// DefaultHumanoidGRUCritic for MI455X (gfx1250, wave32, WMMA, TDM)
//
// Pipeline:
//   k_concat      : build obs[T,352] (f16, zero-padded 341->352)
//   k_cvt_pad x6  : f32 -> f16 weight copies (w_ih0 padded to K=352)
//   k_init        : seed global h0/h1 slots from hidden_states_dn
//   k_gemm_ig0    : IG0[T,1536] = obs @ w_ih0^T + b_ih0   (v_wmma_f32_16x16x32_f16)
//   k_recurrent   : 16 cooperating WGs, LDS-resident W_hh0/W_ih1/W_hh1 slices
//                   (filled via tensor_load_to_lds with LDS padding),
//                   2 grid syncs per step (cluster barrier + atomic barrier),
//                   h broadcast via global_load_async_to_lds_b32
//   k_mlp         : y[T] = mw2 @ relu(mw1 @ relu(mw0 @ h1 + mb0) + mb1) + mb2
// ---------------------------------------------------------------------------

#define T_STEPS 16384
#define IN_F    341
#define IN_P    352     // padded K (11 * 32)
#define HGRU    512
#define G3      1536    // 3*H
#define NWG     16      // workgroups in sequential kernel (cluster-sized)
#define RTHREADS 512

typedef __attribute__((ext_vector_type(16))) _Float16 v16h;
typedef __attribute__((ext_vector_type(8)))  float    v8f;

#if defined(__has_builtin)
#  if __has_builtin(__builtin_amdgcn_tensor_load_to_lds) && __has_builtin(__builtin_amdgcn_s_wait_tensorcnt)
#    define HAVE_TDM 1
#  endif
#  if __has_builtin(__builtin_amdgcn_global_load_async_to_lds_b32) && __has_builtin(__builtin_amdgcn_s_wait_asynccnt)
#    define HAVE_ASYNC_LDS 1
#  endif
#endif
#ifndef HAVE_TDM
#define HAVE_TDM 0
#endif
#ifndef HAVE_ASYNC_LDS
#define HAVE_ASYNC_LDS 0
#endif

__device__ __forceinline__ float sigm(float x) { return 1.0f / (1.0f + __expf(-x)); }
__device__ __forceinline__ float h2f(unsigned short s) {
  _Float16 h = __builtin_bit_cast(_Float16, s);
  return (float)h;
}

// ---------------------------------------------------------------------------
// obs concatenation -> f16 padded
// ---------------------------------------------------------------------------
struct ObsPtrs { const float* p[13]; };

__constant__ int kSegStart[14] = {0, 21, 42, 202, 298, 301, 304, 325, 328, 332, 335, 338, 340, 341};

__global__ __launch_bounds__(256) void k_concat(ObsPtrs ptrs, _Float16* __restrict__ obs) {
  int idx = blockIdx.x * 256 + threadIdx.x;
  if (idx >= T_STEPS * IN_P) return;
  int t = idx / IN_P, c = idx % IN_P;
  float v = 0.0f;
  if (c < IN_F) {
    int s = 0;
    while (c >= kSegStart[s + 1]) ++s;
    int w = kSegStart[s + 1] - kSegStart[s];
    v = ptrs.p[s][t * w + (c - kSegStart[s])];
  }
  obs[idx] = (_Float16)v;
}

// ---------------------------------------------------------------------------
// f32 -> f16 convert with optional column padding
// ---------------------------------------------------------------------------
__global__ __launch_bounds__(256) void k_cvt_pad(const float* __restrict__ src,
                                                 _Float16* __restrict__ dst,
                                                 int rows, int scols, int dcols) {
  int idx = blockIdx.x * 256 + threadIdx.x;
  if (idx >= rows * dcols) return;
  int r = idx / dcols, c = idx % dcols;
  dst[idx] = (_Float16)(c < scols ? src[r * scols + c] : 0.0f);
}

// ---------------------------------------------------------------------------
// seed h slots
// ---------------------------------------------------------------------------
__global__ __launch_bounds__(256) void k_init(const float* __restrict__ hid,
                                              float* __restrict__ g_h0,
                                              float* __restrict__ g_h1) {
  int i = blockIdx.x * 256 + threadIdx.x;
  if (i < HGRU)            g_h0[i] = hid[i];
  else if (i < 2 * HGRU)   g_h1[i - HGRU] = hid[i];
}

// ---------------------------------------------------------------------------
// IG0 = obs @ w_ih0^T + b_ih0     (WMMA 16x16x32 f16 -> f32)
// ---------------------------------------------------------------------------
__global__ __launch_bounds__(256) void k_gemm_ig0(const _Float16* __restrict__ obs,
                                                  const _Float16* __restrict__ w,
                                                  const float* __restrict__ bias,
                                                  float* __restrict__ ig0) {
  int wv   = (blockIdx.x * 256 + threadIdx.x) >> 5;
  int lane = threadIdx.x & 31;
  int mtile = wv % (T_STEPS / 16);
  int ngrp  = wv / (T_STEPS / 16);   // 0..23
  int lm   = lane & 15;
  int koff = (lane >> 4) * 16;
  int n0   = ngrp * 64;

  const _Float16* aptr = obs + (size_t)(mtile * 16 + lm) * IN_P + koff;
  const _Float16* bptr = w   + (size_t)(n0 + lm) * IN_P + koff;

  v8f acc0 = {}, acc1 = {}, acc2 = {}, acc3 = {};
#pragma unroll
  for (int kk = 0; kk < IN_P / 32; ++kk) {
    v16h a  = *(const v16h*)(aptr + kk * 32);
    v16h b0 = *(const v16h*)(bptr + kk * 32);
    v16h b1 = *(const v16h*)(bptr + 16 * IN_P + kk * 32);
    v16h b2 = *(const v16h*)(bptr + 32 * IN_P + kk * 32);
    v16h b3 = *(const v16h*)(bptr + 48 * IN_P + kk * 32);
    acc0 = __builtin_amdgcn_wmma_f32_16x16x32_f16(false, a, false, b0, (short)0, acc0, false, false);
    acc1 = __builtin_amdgcn_wmma_f32_16x16x32_f16(false, a, false, b1, (short)0, acc1, false, false);
    acc2 = __builtin_amdgcn_wmma_f32_16x16x32_f16(false, a, false, b2, (short)0, acc2, false, false);
    acc3 = __builtin_amdgcn_wmma_f32_16x16x32_f16(false, a, false, b3, (short)0, acc3, false, false);
  }

  int col   = n0 + lm;
  int rbase = mtile * 16 + ((lane >> 4) ? 8 : 0);
  float bi0 = bias[col], bi1 = bias[col + 16], bi2 = bias[col + 32], bi3 = bias[col + 48];
#pragma unroll
  for (int r = 0; r < 8; ++r) {
    size_t o = (size_t)(rbase + r) * G3;
    ig0[o + col]      = acc0[r] + bi0;
    ig0[o + col + 16] = acc1[r] + bi1;
    ig0[o + col + 32] = acc2[r] + bi2;
    ig0[o + col + 48] = acc3[r] + bi3;
  }
}

// ---------------------------------------------------------------------------
// sequential GRU recurrence: 16 WGs, LDS-resident weight slices
// ---------------------------------------------------------------------------
#define WROW_U   260                 // 256 u32 (512 f16) + 4 u32 pad (bank skew)
#define WSLICE_U (96 * WROW_U)       // one 96-row slice
#define U_H0     (3 * WSLICE_U)
#define U_H1     (U_H0 + HGRU)
#define U_HG     (U_H1 + HGRU)       // 192 floats (hg0 | hg1)
#define U_IG1    (U_HG + 192)        // 96 floats
#define SMEM_U32 (U_IG1 + 96)
#define SMEM_BYTES (SMEM_U32 * 4)    // 304,768 B  (< 320 KB WGP LDS)

#if HAVE_TDM
typedef unsigned tdm_v4u __attribute__((ext_vector_type(4)));
typedef int      tdm_v8i __attribute__((ext_vector_type(8)));
typedef int      tdm_v4i __attribute__((ext_vector_type(4)));

// TDM: DMA one 32-row x 512-col f16 tile from global into LDS, inserting the
// 16B bank-skew pad after every 1024B row (pad_interval=7 -> 256 DWORDs,
// pad_amount=3 -> 4 DWORDs).
__device__ __forceinline__ void tdm_load_32x512_f16(unsigned lds_addr, const void* gsrc) {
  unsigned long long ga = (unsigned long long)(size_t)gsrc;
  tdm_v4u g0;
  g0[0] = 1u;                                                   // count=1 (user D#)
  g0[1] = lds_addr;                                             // lds_addr [63:32]
  g0[2] = (unsigned)ga;                                         // global_addr lo
  g0[3] = (unsigned)((ga >> 32) & 0x01FFFFFFull) | (2u << 30);  // addr hi | type=2
  tdm_v8i g1;
  g1[0] = (int)((1u << 16) |        // data_size = 2 bytes
                (1u << 20) |        // pad_enable
                (7u << 22) |        // pad_interval = 256 DWORDs
                (3u << 25));        // pad_amount   = 4 DWORDs
  g1[1] = (int)(512u << 16);        // tensor_dim0[15:0]  (=512)
  g1[2] = (int)(1536u << 16);       // tensor_dim0 hi=0 | tensor_dim1 lo (=1536)
  g1[3] = (int)(512u << 16);        // tensor_dim1 hi=0 | tile_dim0 (=512)
  g1[4] = 32;                       // tile_dim1=32, tile_dim2=0
  g1[5] = 512;                      // tensor_dim0_stride lo
  g1[6] = 0;                        // stride hi | dim1_stride lo
  g1[7] = 0;
  tdm_v4i gz = {0, 0, 0, 0};
#if __clang_major__ >= 23
  tdm_v8i gz8 = {0, 0, 0, 0, 0, 0, 0, 0};
  __builtin_amdgcn_tensor_load_to_lds(g0, g1, gz, gz, gz8, 0);
#else
  __builtin_amdgcn_tensor_load_to_lds(g0, g1, gz, gz, 0);
#endif
}
#endif  // HAVE_TDM

__device__ __forceinline__ void grid_sync(unsigned* cnt, unsigned epoch) {
  __syncthreads();
  // CDNA5 cluster barrier: real sync when dispatched as a 16-WG cluster,
  // architected NOP otherwise (ClusterID==0). Atomic barrier below is the
  // dispatch-agnostic fallback that is always correct.
  __builtin_amdgcn_s_cluster_barrier();
  if (threadIdx.x == 0) {
    __hip_atomic_fetch_add(cnt, 1u, __ATOMIC_ACQ_REL, __HIP_MEMORY_SCOPE_AGENT);
    while (__hip_atomic_load(cnt, __ATOMIC_ACQUIRE, __HIP_MEMORY_SCOPE_AGENT) < epoch * NWG) {
      __builtin_amdgcn_s_sleep(2);
    }
  }
  __syncthreads();
}

// broadcast-refresh one h vector (global f32[512]) into LDS staging
__device__ __forceinline__ void refresh_h(const float* __restrict__ gsrc,
                                          float* __restrict__ lds_dst, int tid) {
#if HAVE_ASYNC_LDS
  for (int i = tid; i < HGRU; i += RTHREADS) {
    __builtin_amdgcn_global_load_async_to_lds_b32(
        (int*)(gsrc + i), (int*)(lds_dst + i),
        0, /*cpol: scope=DEV*/ (2 << 3));
  }
  __builtin_amdgcn_s_wait_asynccnt(0);
#else
  for (int i = tid; i < HGRU; i += RTHREADS)
    lds_dst[i] = __hip_atomic_load(&gsrc[i], __ATOMIC_ACQUIRE, __HIP_MEMORY_SCOPE_AGENT);
#endif
}

__global__ __launch_bounds__(RTHREADS, 1) void k_recurrent(
    const _Float16* __restrict__ whh0, const _Float16* __restrict__ wih1,
    const _Float16* __restrict__ whh1, const float* __restrict__ ig0,
    const float* __restrict__ b_n0, const float* __restrict__ b_ih1,
    const float* __restrict__ b_n1, float* __restrict__ g_h0,
    float* __restrict__ g_h1, _Float16* __restrict__ h1f,
    unsigned* __restrict__ cnt, float* __restrict__ out) {
  extern __shared__ unsigned smem_u[];
  float* smem_f = (float*)smem_u;

  const int wg = blockIdx.x;
  const int tid = threadIdx.x;

  // ---- fill LDS weight slices: rows {g*512 + wg*32 + i} for g=0..2, i=0..31
#if HAVE_TDM
  if (tid == 0) {
    unsigned lds0 = __builtin_amdgcn_groupstaticsize();  // dynamic-LDS base offset
    for (int m = 0; m < 3; ++m) {
      const _Float16* src = (m == 0) ? whh0 : (m == 1) ? wih1 : whh1;
      for (int g = 0; g < 3; ++g) {
        unsigned lds_addr = lds0 + (unsigned)(m * WSLICE_U + (g * 32) * WROW_U) * 4u;
        tdm_load_32x512_f16(lds_addr, src + (size_t)(g * HGRU + wg * 32) * HGRU);
      }
    }
  }
  __builtin_amdgcn_s_wait_tensorcnt(0);
#else
  for (int m = 0; m < 3; ++m) {
    const unsigned* srcu = (const unsigned*)((m == 0) ? whh0 : (m == 1) ? wih1 : whh1);
    for (int idx = tid; idx < 96 * 256; idx += RTHREADS) {
      int lr = idx >> 8;          // local row 0..95
      int k  = idx & 255;         // u32 within row
      int g  = lr >> 5, il = lr & 31;
      int grow = g * HGRU + wg * 32 + il;
      smem_u[m * WSLICE_U + lr * WROW_U + k] = srcu[grow * 256 + k];
    }
  }
#endif
  for (int i = tid; i < HGRU; i += RTHREADS) {
    smem_f[U_H0 + i] = __hip_atomic_load(&g_h0[i], __ATOMIC_RELAXED, __HIP_MEMORY_SCOPE_AGENT);
    smem_f[U_H1 + i] = __hip_atomic_load(&g_h1[i], __ATOMIC_RELAXED, __HIP_MEMORY_SCOPE_AGENT);
  }
  __syncthreads();

  unsigned epoch = 0;
  const int r2 = tid >> 1, half = tid & 1;
  const int r4 = tid >> 2, q = tid & 3;

  for (int t = 0; t < T_STEPS; ++t) {
    // ---- phase 1: hg0 = W_hh0 @ h0_prev (rows 0..95), hg1 = W_hh1 @ h1_prev (96..191)
    if (r2 < 192) {
      int mat = (r2 < 96) ? 0 : 2;
      int lr  = (r2 < 96) ? r2 : r2 - 96;
      const unsigned* wrow = &smem_u[mat * WSLICE_U + lr * WROW_U + half * 128];
      const float* hv = &smem_f[((mat == 0) ? U_H0 : U_H1) + half * 256];
      float s = 0.0f;
#pragma unroll 4
      for (int j = 0; j < 128; ++j) {
        unsigned u = wrow[j];
        s = fmaf(h2f((unsigned short)u),         hv[2 * j],     s);
        s = fmaf(h2f((unsigned short)(u >> 16)), hv[2 * j + 1], s);
      }
      s += __shfl_xor(s, 1);
      if (half == 0) smem_f[U_HG + r2] = s;
    }
    __syncthreads();

    // ---- layer-0 combine (32 hidden elems owned by this WG)
    if (tid < 32) {
      int gi = wg * 32 + tid;
      size_t o = (size_t)t * G3;
      float ir  = ig0[o + gi];
      float iz  = ig0[o + HGRU + gi];
      float inn = ig0[o + 2 * HGRU + gi];
      if (t + 1 < T_STEPS) {   // prefetch next step's IG0 rows (global_prefetch_b8)
        __builtin_prefetch(&ig0[o + G3 + gi], 0, 1);
        __builtin_prefetch(&ig0[o + G3 + HGRU + gi], 0, 1);
        __builtin_prefetch(&ig0[o + G3 + 2 * HGRU + gi], 0, 1);
      }
      float hr = smem_f[U_HG + tid];
      float hz = smem_f[U_HG + 32 + tid];
      float hn = smem_f[U_HG + 64 + tid];
      float r = sigm(ir + hr), z = sigm(iz + hz);
      float n = tanhf(inn + r * (hn + b_n0[gi]));
      float h0p = smem_f[U_H0 + gi];
      float h0n = n + z * (h0p - n);
      __hip_atomic_store(&g_h0[gi], h0n, __ATOMIC_RELEASE, __HIP_MEMORY_SCOPE_AGENT);
      if (t == T_STEPS - 1) out[T_STEPS + gi] = h0n;
    }
    grid_sync(cnt, ++epoch);
    refresh_h(g_h0, &smem_f[U_H0], tid);
    __syncthreads();

    // ---- phase 2: ig1 = W_ih1 @ h0_new (rows 0..95, 4 threads per row)
    if (r4 < 96) {
      const unsigned* wrow = &smem_u[1 * WSLICE_U + r4 * WROW_U + q * 64];
      const float* hv = &smem_f[U_H0 + q * 128];
      float s = 0.0f;
#pragma unroll 4
      for (int j = 0; j < 64; ++j) {
        unsigned u = wrow[j];
        s = fmaf(h2f((unsigned short)u),         hv[2 * j],     s);
        s = fmaf(h2f((unsigned short)(u >> 16)), hv[2 * j + 1], s);
      }
      s += __shfl_xor(s, 1);
      s += __shfl_xor(s, 2);
      if (q == 0) smem_f[U_IG1 + r4] = s;
    }
    __syncthreads();

    // ---- layer-1 combine
    if (tid < 32) {
      int gi = wg * 32 + tid;
      float ir  = smem_f[U_IG1 + tid]      + b_ih1[gi];
      float iz  = smem_f[U_IG1 + 32 + tid] + b_ih1[HGRU + gi];
      float inn = smem_f[U_IG1 + 64 + tid] + b_ih1[2 * HGRU + gi];
      float hr = smem_f[U_HG + 96 + tid];
      float hz = smem_f[U_HG + 128 + tid];
      float hn = smem_f[U_HG + 160 + tid];
      float r = sigm(ir + hr), z = sigm(iz + hz);
      float n = tanhf(inn + r * (hn + b_n1[gi]));
      float h1p = smem_f[U_H1 + gi];
      float h1n = n + z * (h1p - n);
      __hip_atomic_store(&g_h1[gi], h1n, __ATOMIC_RELEASE, __HIP_MEMORY_SCOPE_AGENT);
      h1f[(size_t)t * HGRU + gi] = (_Float16)h1n;
      if (t == T_STEPS - 1) out[T_STEPS + HGRU + gi] = h1n;
    }
    grid_sync(cnt, ++epoch);
    refresh_h(g_h1, &smem_f[U_H1], tid);
    __syncthreads();
  }
}

// ---------------------------------------------------------------------------
// MLP head over all T (time-parallel): two WMMA stages + VALU final dot
// ---------------------------------------------------------------------------
__global__ __launch_bounds__(256) void k_mlp(const _Float16* __restrict__ h1f,
                                             const _Float16* __restrict__ mw0f,
                                             const float* __restrict__ mb0,
                                             const _Float16* __restrict__ mw1f,
                                             const float* __restrict__ mb1,
                                             const float* __restrict__ mw2,
                                             const float* __restrict__ mb2,
                                             float* __restrict__ out) {
  __shared__ _Float16 vbuf[8][16 * 64];
  __shared__ float    v2buf[8][16 * 64];
  int wid = threadIdx.x >> 5, lane = threadIdx.x & 31;
  int mtile = blockIdx.x * 8 + wid;
  int t0 = mtile * 16;
  int lm = lane & 15, kh = (lane >> 4) * 16;
  int rbase = (lane >> 4) ? 8 : 0;

  // stage 0: [16,512] @ [512,64]^T + mb0, relu
  v8f acc[4];
#pragma unroll
  for (int j = 0; j < 4; ++j) {
    float b = mb0[j * 16 + lm];
    v8f tv; for (int r = 0; r < 8; ++r) tv[r] = b;
    acc[j] = tv;
  }
  const _Float16* ap = h1f + (size_t)(t0 + lm) * HGRU + kh;
#pragma unroll 4
  for (int kk = 0; kk < HGRU / 32; ++kk) {
    v16h a = *(const v16h*)(ap + kk * 32);
#pragma unroll
    for (int j = 0; j < 4; ++j) {
      v16h b = *(const v16h*)(mw0f + (size_t)(j * 16 + lm) * HGRU + kh + kk * 32);
      acc[j] = __builtin_amdgcn_wmma_f32_16x16x32_f16(false, a, false, b, (short)0, acc[j], false, false);
    }
  }
#pragma unroll
  for (int j = 0; j < 4; ++j)
#pragma unroll
    for (int r = 0; r < 8; ++r) {
      float v = acc[j][r];
      vbuf[wid][(rbase + r) * 64 + j * 16 + lm] = (_Float16)(v > 0.0f ? v : 0.0f);
    }
  __syncthreads();

  // stage 1: [16,64] @ [64,64]^T + mb1, relu
  v8f acc2[4];
#pragma unroll
  for (int j = 0; j < 4; ++j) {
    float b = mb1[j * 16 + lm];
    v8f tv; for (int r = 0; r < 8; ++r) tv[r] = b;
    acc2[j] = tv;
  }
#pragma unroll
  for (int kk = 0; kk < 2; ++kk) {
    v16h a = *(const v16h*)(&vbuf[wid][lm * 64 + kh + kk * 32]);
#pragma unroll
    for (int j = 0; j < 4; ++j) {
      v16h b = *(const v16h*)(mw1f + (size_t)(j * 16 + lm) * 64 + kh + kk * 32);
      acc2[j] = __builtin_amdgcn_wmma_f32_16x16x32_f16(false, a, false, b, (short)0, acc2[j], false, false);
    }
  }
#pragma unroll
  for (int j = 0; j < 4; ++j)
#pragma unroll
    for (int r = 0; r < 8; ++r) {
      float v = acc2[j][r];
      v2buf[wid][(rbase + r) * 64 + j * 16 + lm] = (v > 0.0f ? v : 0.0f);
    }
  __syncthreads();

  // final: y = mw2 . v2 + mb2   (one row per lane 0..15)
  if (lane < 16) {
    float s = mb2[0];
#pragma unroll 8
    for (int c = 0; c < 64; ++c) s = fmaf(v2buf[wid][lane * 64 + c], mw2[c], s);
    out[t0 + lane] = s;
  }
}

// ---------------------------------------------------------------------------
// host launcher
// ---------------------------------------------------------------------------
extern "C" void kernel_launch(void* const* d_in, const int* in_sizes, int n_in,
                              void* d_out, int out_size, void* d_ws, size_t ws_size,
                              hipStream_t stream) {
  (void)in_sizes; (void)n_in; (void)out_size; (void)ws_size;

  char* ws = (char*)d_ws;
  size_t off = 0;
  auto take = [&](size_t bytes) {
    size_t o = off;
    off += (bytes + 255) & ~(size_t)255;
    return o;
  };

  unsigned* cnt      = (unsigned*)(ws + take(4096));
  float*    g_h0     = (float*)(ws + take(HGRU * 4));
  float*    g_h1     = (float*)(ws + take(HGRU * 4));
  _Float16* obs      = (_Float16*)(ws + take((size_t)T_STEPS * IN_P * 2));
  _Float16* wih0_h   = (_Float16*)(ws + take((size_t)G3 * IN_P * 2));
  _Float16* whh0_h   = (_Float16*)(ws + take((size_t)G3 * HGRU * 2));
  _Float16* wih1_h   = (_Float16*)(ws + take((size_t)G3 * HGRU * 2));
  _Float16* whh1_h   = (_Float16*)(ws + take((size_t)G3 * HGRU * 2));
  _Float16* mw0_h    = (_Float16*)(ws + take((size_t)64 * HGRU * 2));
  _Float16* mw1_h    = (_Float16*)(ws + take((size_t)64 * 64 * 2));
  _Float16* h1f      = (_Float16*)(ws + take((size_t)T_STEPS * HGRU * 2));
  float*    ig0      = (float*)(ws + take((size_t)T_STEPS * G3 * 4));

  const float* hid   = (const float*)d_in[13];
  const float* w_ih0 = (const float*)d_in[14];
  const float* w_hh0 = (const float*)d_in[15];
  const float* b_ih0 = (const float*)d_in[16];
  const float* b_n0  = (const float*)d_in[17];
  const float* w_ih1 = (const float*)d_in[18];
  const float* w_hh1 = (const float*)d_in[19];
  const float* b_ih1 = (const float*)d_in[20];
  const float* b_n1  = (const float*)d_in[21];
  const float* mw0   = (const float*)d_in[22];
  const float* mb0   = (const float*)d_in[23];
  const float* mw1   = (const float*)d_in[24];
  const float* mb1   = (const float*)d_in[25];
  const float* mw2   = (const float*)d_in[26];
  const float* mb2   = (const float*)d_in[27];
  float* out = (float*)d_out;

  (void)hipMemsetAsync(cnt, 0, 4096, stream);

  ObsPtrs op;
  for (int i = 0; i < 13; ++i) op.p[i] = (const float*)d_in[i];
  k_concat<<<(T_STEPS * IN_P + 255) / 256, 256, 0, stream>>>(op, obs);

  k_cvt_pad<<<(G3 * IN_P + 255) / 256, 256, 0, stream>>>(w_ih0, wih0_h, G3, IN_F, IN_P);
  k_cvt_pad<<<(G3 * HGRU + 255) / 256, 256, 0, stream>>>(w_hh0, whh0_h, G3, HGRU, HGRU);
  k_cvt_pad<<<(G3 * HGRU + 255) / 256, 256, 0, stream>>>(w_ih1, wih1_h, G3, HGRU, HGRU);
  k_cvt_pad<<<(G3 * HGRU + 255) / 256, 256, 0, stream>>>(w_hh1, whh1_h, G3, HGRU, HGRU);
  k_cvt_pad<<<(64 * HGRU + 255) / 256, 256, 0, stream>>>(mw0, mw0_h, 64, HGRU, HGRU);
  k_cvt_pad<<<(64 * 64 + 255) / 256, 256, 0, stream>>>(mw1, mw1_h, 64, 64, 64);

  k_init<<<(2 * HGRU + 255) / 256, 256, 0, stream>>>(hid, g_h0, g_h1);

  // IG0 GEMM: 1024 M-tiles x 24 N-groups = 24576 waves = 3072 blocks
  k_gemm_ig0<<<(T_STEPS / 16) * (G3 / 64) / 8, 256, 0, stream>>>(obs, wih0_h, b_ih0, ig0);

  (void)hipFuncSetAttribute((const void*)k_recurrent,
                            hipFuncAttributeMaxDynamicSharedMemorySize, SMEM_BYTES);
  k_recurrent<<<NWG, RTHREADS, SMEM_BYTES, stream>>>(whh0_h, wih1_h, whh1_h, ig0,
                                                     b_n0, b_ih1, b_n1, g_h0, g_h1,
                                                     h1f, cnt, out);

  k_mlp<<<T_STEPS / 16 / 8, 256, 0, stream>>>(h1f, mw0_h, mb0, mw1_h, mb1, mw2, mb2, out);
}
